// MDHG_70111046140136
// MI455X (gfx1250) — matrix operand order repaired and checked by reading, not compiled, not run on hardware
//
#include <hip/hip_runtime.h>
#include <hip/hip_bf16.h>
#include <math.h>

#define NNODE 100000
#define DIM   100
#define KCL   100
#define LAY   3
#define NNZE  1600000
#define DP    128   // padded dim (4 k-chunks of 32 for bf16 WMMA)

typedef __attribute__((ext_vector_type(16))) __bf16 v16bf;
typedef __attribute__((ext_vector_type(8)))  float  v8f;

union V16 { uint4 u[2]; v16bf v; };

__device__ __forceinline__ unsigned short f2bf(float f) {
    unsigned int u = __float_as_uint(f);
    u += 0x7FFFu + ((u >> 16) & 1u);   // round-to-nearest-even
    return (unsigned short)(u >> 16);
}

__device__ __forceinline__ float warpMax(float v) {
    for (int m = 16; m > 0; m >>= 1) v = fmaxf(v, __shfl_xor(v, m, 32));
    return v;
}
__device__ __forceinline__ float warpSum(float v) {
    for (int m = 16; m > 0; m >>= 1) v += __shfl_xor(v, m, 32);
    return v;
}

// ---------------- weight pack: W[k,cols] f32 -> Bt[n*DP+k] bf16 (transposed, zero-padded)
__global__ void pack_weight_k(const float* __restrict__ src, unsigned short* __restrict__ dst,
                              int rows, int cols) {
    int i = blockIdx.x * blockDim.x + threadIdx.x;
    if (i >= DP * DP) return;
    int n = i >> 7, k = i & 127;
    dst[i] = (n < cols && k < rows) ? f2bf(src[k * cols + n]) : (unsigned short)0;
}

// ---------------- init: sumF = embedding, sumH = 0, xb = bf16(embedding) padded
__global__ void init_k(const float* __restrict__ emb, unsigned short* __restrict__ xb,
                       float* __restrict__ sumF, float* __restrict__ sumH) {
    for (int i = blockIdx.x * blockDim.x + threadIdx.x; i < NNODE * DP;
         i += gridDim.x * blockDim.x) {
        int n = i >> 7, c = i & 127;
        if (c < DIM) {
            float v = emb[n * DIM + c];
            xb[i] = f2bf(v);
            sumF[n * DIM + c] = v;
            sumH[n * DIM + c] = 0.f;
        } else {
            xb[i] = 0;
        }
    }
}

__global__ void zero_k(float* __restrict__ p, int n) {
    for (int i = blockIdx.x * blockDim.x + threadIdx.x; i < n; i += gridDim.x * blockDim.x)
        p[i] = 0.f;
}

// ---------------- WMMA GEMM: C[M,ncols] = A[M,DP](bf16) x B[DP,ncols], Bt in [n][k] layout
__global__ __launch_bounds__(256) void gemm_bf16_k(
    const unsigned short* __restrict__ A, const unsigned short* __restrict__ Bt,
    float* __restrict__ C, int M, int ncols) {
    __shared__ __attribute__((aligned(32))) unsigned short ldsB[DP * DP];  // 32 KB
    int tid = threadIdx.x;
    {   // cooperative load of whole B (transposed) into LDS
        const uint4* s = (const uint4*)Bt;
        uint4* d = (uint4*)ldsB;
        for (int i = tid; i < DP * DP / 8; i += 256) d[i] = s[i];
    }
    __syncthreads();
    int wave = tid >> 5, lane = tid & 31;
    int rlo = lane & 15, hi = lane >> 4;
    int koff8 = hi * 8;    // unused; A loads use khalf (contiguous k = 16*hi + s)
    int khalf = hi * 16;   // B-fragment K-half
    int row0 = blockIdx.x * 128 + wave * 16;
    int arow = row0 + rlo; if (arow > M - 1) arow = M - 1;   // clamp loads, guard stores
    const unsigned short* Ap = A + (size_t)arow * DP;

    v8f acc[8];
    v8f zv = {0.f, 0.f, 0.f, 0.f, 0.f, 0.f, 0.f, 0.f};
#pragma unroll
    for (int n = 0; n < 8; ++n) acc[n] = zv;

#pragma unroll
    for (int kt = 0; kt < 4; ++kt) {
        V16 a;
        a.u[0] = *(const uint4*)(Ap + kt * 32 + khalf);
        a.u[1] = *(const uint4*)(Ap + kt * 32 + khalf + 8);
#pragma unroll
        for (int n = 0; n < 8; ++n) {
            v16bf b = *(const v16bf*)(&ldsB[(n * 16 + rlo) * DP + kt * 32 + khalf]);
            acc[n] = __builtin_amdgcn_wmma_f32_16x16x32_bf16(
                false, a.v, false, b, (short)0, acc[n], false, false);
        }
    }
#pragma unroll
    for (int n = 0; n < 8; ++n) {
        int col = n * 16 + rlo;
        if (col >= ncols) continue;
#pragma unroll
        for (int r = 0; r < 8; ++r) {
            int row = row0 + r + hi * 8;
            if (row < M) C[(size_t)row * ncols + col] = acc[n][r];
        }
    }
}

// ---------------- edge-parallel SpMM: y[r,:] += v * x[c,:]  (wave per edge)
__global__ void spmm_k(const int* __restrict__ er, const int* __restrict__ ec,
                       const float* __restrict__ ev, const float* __restrict__ xs,
                       float* __restrict__ y) {
    int gt = blockIdx.x * blockDim.x + threadIdx.x;
    int wid = gt >> 5, lane = gt & 31;
    int nw = (gridDim.x * blockDim.x) >> 5;
    for (int e = wid; e < NNZE; e += nw) {
        int r = er[e], c = ec[e];
        float v = ev[e];
        const float* xc = xs + (size_t)c * DIM;
        float* yr = y + (size_t)r * DIM;
        for (int d = lane; d < DIM; d += 32) atomicAdd(&yr[d], v * xc[d]);
    }
}

// ---------------- f32 [N,DIM] -> bf16 [N,DP] padded
__global__ void pad_bf16_k(const float* __restrict__ s, unsigned short* __restrict__ d) {
    for (int i = blockIdx.x * blockDim.x + threadIdx.x; i < NNODE * DP;
         i += gridDim.x * blockDim.x) {
        int n = i >> 7, c = i & 127;
        d[i] = (c < DIM) ? f2bf(s[n * DIM + c]) : (unsigned short)0;
    }
}

// ---------------- H1pre = relu(t + y) -> bf16 padded
__global__ void relures_k(const float* __restrict__ t, const float* __restrict__ y,
                          unsigned short* __restrict__ out) {
    for (int i = blockIdx.x * blockDim.x + threadIdx.x; i < NNODE * DP;
         i += gridDim.x * blockDim.x) {
        int n = i >> 7, c = i & 127;
        out[i] = (c < DIM) ? f2bf(fmaxf(t[n * DIM + c] + y[n * DIM + c], 0.f))
                           : (unsigned short)0;
    }
}

// ---------------- row softmax + adj column-normalization (wave per row)
__global__ __launch_bounds__(256) void softmax_k(
    const float* __restrict__ z, const float* __restrict__ adj,
    unsigned short* __restrict__ H1b, unsigned short* __restrict__ Hwb) {
    int wave = threadIdx.x >> 5, lane = threadIdx.x & 31;
    int row = blockIdx.x * 8 + wave;
    if (row >= NNODE) return;
    const float* zr = z + (size_t)row * KCL;
    float v[4];
    float m = -3.4e38f;
#pragma unroll
    for (int j = 0; j < 4; ++j) {
        int d = lane + 32 * j;
        v[j] = (d < KCL) ? zr[d] : -3.4e38f;
        m = fmaxf(m, v[j]);
    }
    m = warpMax(m);
    float s = 0.f;
#pragma unroll
    for (int j = 0; j < 4; ++j) {
        int d = lane + 32 * j;
        v[j] = (d < KCL) ? __expf(v[j] - m) : 0.f;
        s += v[j];
    }
    s = warpSum(s);
    float inv = 1.f / s;
    float s2 = 0.f;
#pragma unroll
    for (int j = 0; j < 4; ++j) { v[j] *= inv; s2 += v[j]; }
    s2 = warpSum(s2);                       // = sum_k H1 (≈1)
    float a = adj[row];
    float invw = 1.f / (a * s2);            // h/(sum over clusters) per reference
    unsigned short* h1 = H1b + (size_t)row * DP;
    unsigned short* hw = Hwb + (size_t)row * DP;
#pragma unroll
    for (int j = 0; j < 4; ++j) {
        int d = lane + 32 * j;
        if (d < KCL) { h1[d] = f2bf(v[j]); hw[d] = f2bf(v[j] * a * invw); }
        else if (d < DP) { h1[d] = 0; hw[d] = 0; }
    }
}

// ---------------- WMMA reduction: Hout[k,d] = sum_n Hw[n,k] * X[n,d]   (atomic partials)
__global__ __launch_bounds__(256) void hred_k(
    const unsigned short* __restrict__ Hwb, const unsigned short* __restrict__ Xb,
    float* __restrict__ Hout) {
    __shared__ __attribute__((aligned(32))) unsigned short ldsA[DP * 32];  // [cluster][node]
    __shared__ __attribute__((aligned(32))) unsigned short ldsX[DP * 32];  // [d][node]
    int tid = threadIdx.x, wave = tid >> 5, lane = tid & 31;
    int rlo = lane & 15, hi = lane >> 4;
    int koff8 = hi * 8, khalf = hi * 16;

    v8f acc[8];
    v8f zv = {0.f, 0.f, 0.f, 0.f, 0.f, 0.f, 0.f, 0.f};
#pragma unroll
    for (int n = 0; n < 8; ++n) acc[n] = zv;

    const int chunks = NNODE / 32;  // 3125 exact
    for (int c = blockIdx.x; c < chunks; c += gridDim.x) {
        int base = c * 32;
        for (int e = tid; e < 32 * DP; e += 256) {  // stage transposed tiles
            int node = e >> 7, col = e & 127;
            ldsA[col * 32 + node] = Hwb[(size_t)(base + node) * DP + col];
            ldsX[col * 32 + node] = Xb[(size_t)(base + node) * DP + col];
        }
        __syncthreads();
        V16 a;
        int arow = wave * 16 + rlo;
        a.u[0] = *(const uint4*)(&ldsA[arow * 32 + khalf]);
        a.u[1] = *(const uint4*)(&ldsA[arow * 32 + khalf + 8]);
#pragma unroll
        for (int n = 0; n < 8; ++n) {
            v16bf b = *(const v16bf*)(&ldsX[(n * 16 + rlo) * 32 + khalf]);
            acc[n] = __builtin_amdgcn_wmma_f32_16x16x32_bf16(
                false, a.v, false, b, (short)0, acc[n], false, false);
        }
        __syncthreads();
    }
#pragma unroll
    for (int n = 0; n < 8; ++n)
#pragma unroll
        for (int r = 0; r < 8; ++r) {
            int krow = wave * 16 + r + hi * 8;
            atomicAdd(&Hout[krow * DP + n * 16 + rlo], acc[n][r]);
        }
}

// ---------------- h [128,128] f32 -> Bt layout bf16: ht[d*DP + k] = h[k*DP + d]
__global__ void packh_k(const float* __restrict__ h, unsigned short* __restrict__ ht) {
    int i = blockIdx.x * blockDim.x + threadIdx.x;
    if (i >= DP * DP) return;
    int d = i >> 7, k = i & 127;
    ht[i] = f2bf(h[k * DP + d]);
}

// ---------------- x_next = hn + y; accumulate l2norm(x_next), l2norm(hn); write xb
__global__ __launch_bounds__(256) void finalize_k(
    const float* __restrict__ hn, const float* __restrict__ y,
    float* __restrict__ sumF, float* __restrict__ sumH, unsigned short* __restrict__ xb) {
    int wave = threadIdx.x >> 5, lane = threadIdx.x & 31;
    int row = blockIdx.x * 8 + wave;
    if (row >= NNODE) return;
    const float* hr = hn + (size_t)row * DIM;
    const float* yr = y + (size_t)row * DIM;
    float hx[4], xx[4], nh = 0.f, nx = 0.f;
#pragma unroll
    for (int j = 0; j < 4; ++j) {
        int d = lane + 32 * j;
        if (d < DIM) {
            hx[j] = hr[d];
            xx[j] = hx[j] + yr[d];
            nh += hx[j] * hx[j];
            nx += xx[j] * xx[j];
        } else { hx[j] = 0.f; xx[j] = 0.f; }
    }
    nh = sqrtf(warpSum(nh));
    nx = sqrtf(warpSum(nx));
    float ih = 1.f / fmaxf(nh, 1e-12f);
    float ix = 1.f / fmaxf(nx, 1e-12f);
#pragma unroll
    for (int j = 0; j < 4; ++j) {
        int d = lane + 32 * j;
        if (d < DIM) {
            sumF[(size_t)row * DIM + d] += xx[j] * ix;
            sumH[(size_t)row * DIM + d] += hx[j] * ih;
            xb[(size_t)row * DP + d] = f2bf(xx[j]);
        } else if (d < DP) {
            xb[(size_t)row * DP + d] = 0;
        }
    }
}

__global__ void scale_k(float* __restrict__ out) {
    for (int i = blockIdx.x * blockDim.x + threadIdx.x; i < NNODE * DIM;
         i += gridDim.x * blockDim.x) {
        out[i] *= 0.25f;                       // / (L+1)
        out[NNODE * DIM + i] *= (1.f / 3.f);   // / L
    }
}

extern "C" void kernel_launch(void* const* d_in, const int* in_sizes, int n_in,
                              void* d_out, int out_size, void* d_ws, size_t ws_size,
                              hipStream_t stream) {
    (void)in_sizes; (void)n_in; (void)out_size; (void)ws_size;
    const float* emb   = (const float*)d_in[0];
    const float* adj   = (const float*)d_in[1];
    const float* evals = (const float*)d_in[2];
    const float* Wit   = (const float*)d_in[3];
    const float* Wi1   = (const float*)d_in[4];
    const float* Wi2   = (const float*)d_in[5];
    const int*   erow  = (const int*)d_in[6];
    const int*   ecol  = (const int*)d_in[7];

    float* out  = (float*)d_out;
    float* sumF = out;
    float* sumH = out + (size_t)NNODE * DIM;

    char* w = (char*)d_ws;
    auto carve = [&](size_t b) -> char* {
        char* r = w; w += (b + 255) & ~(size_t)255; return r;
    };
    unsigned short* xb   = (unsigned short*)carve((size_t)NNODE * DP * 2);  // aliases H1pre
    unsigned short* yb   = (unsigned short*)carve((size_t)NNODE * DP * 2);
    unsigned short* H1b  = (unsigned short*)carve((size_t)NNODE * DP * 2);
    unsigned short* Hwb  = (unsigned short*)carve((size_t)NNODE * DP * 2);
    float*          buf1 = (float*)carve((size_t)NNODE * DIM * 4);
    float*          ybuf = (float*)carve((size_t)NNODE * DIM * 4);
    float*          hsm  = (float*)carve((size_t)DP * DP * 4);
    unsigned short* hbt  = (unsigned short*)carve((size_t)DP * DP * 2);
    unsigned short* WitT = (unsigned short*)carve((size_t)LAY * DP * DP * 2);
    unsigned short* Wi1T = (unsigned short*)carve((size_t)DP * DP * 2);
    unsigned short* Wi2T = (unsigned short*)carve((size_t)DP * DP * 2);

    dim3 b256(256);
    int gw = (DP * DP + 255) / 256;
    for (int i = 0; i < LAY; ++i)
        pack_weight_k<<<gw, b256, 0, stream>>>(Wit + (size_t)i * DIM * DIM,
                                               WitT + (size_t)i * DP * DP, DIM, DIM);
    pack_weight_k<<<gw, b256, 0, stream>>>(Wi1, Wi1T, DIM, DIM);
    pack_weight_k<<<gw, b256, 0, stream>>>(Wi2, Wi2T, DIM, KCL);

    init_k<<<2048, b256, 0, stream>>>(emb, xb, sumF, sumH);

    int gGemm = (NNODE + 127) / 128;
    int gRow  = (NNODE + 7) / 8;
    for (int i = 0; i < LAY; ++i) {
        // x = x @ W_item[i]
        gemm_bf16_k<<<gGemm, b256, 0, stream>>>(xb, WitT + (size_t)i * DP * DP, buf1, NNODE, DIM);
        // x = spmm(adjacency, x)
        zero_k<<<2048, b256, 0, stream>>>(ybuf, NNODE * DIM);
        spmm_k<<<2048, b256, 0, stream>>>(erow, ecol, evals, buf1, ybuf);
        pad_bf16_k<<<2048, b256, 0, stream>>>(ybuf, yb);
        // H1 = relu(x @ W_i1 + x)
        gemm_bf16_k<<<gGemm, b256, 0, stream>>>(yb, Wi1T, buf1, NNODE, DIM);
        relures_k<<<2048, b256, 0, stream>>>(buf1, ybuf, xb);   // H1pre -> xb (free slot)
        // H1 = softmax(H1 @ W_i2)
        gemm_bf16_k<<<gGemm, b256, 0, stream>>>(xb, Wi2T, buf1, NNODE, KCL);
        softmax_k<<<gRow, b256, 0, stream>>>(buf1, adj, H1b, Hwb);
        // h = (normalized H1.T * adj) @ x   -> [K,D]
        zero_k<<<64, b256, 0, stream>>>(hsm, DP * DP);
        hred_k<<<256, b256, 0, stream>>>(Hwb, yb, hsm);
        packh_k<<<gw, b256, 0, stream>>>(hsm, hbt);
        // h = H1 @ h ; x = h + x ; accumulate l2norms
        gemm_bf16_k<<<gGemm, b256, 0, stream>>>(H1b, hbt, buf1, NNODE, DIM);
        finalize_k<<<gRow, b256, 0, stream>>>(buf1, ybuf, sumF, sumH, xb);
    }
    scale_k<<<2048, b256, 0, stream>>>(out);
}
